// KPConvDeformableLayer_13795434955246
// MI455X (gfx1250) — matrix operand order repaired and checked by reading, not compiled, not run on hardware
//
#include <hip/hip_runtime.h>
#include <hip/hip_bf16.h>

typedef _Float16 v16h __attribute__((ext_vector_type(16)));
typedef float    v8f  __attribute__((ext_vector_type(8)));
typedef _Float16 h2t  __attribute__((ext_vector_type(2)));

namespace {
constexpr int NPTS = 50000;
constexpr int MNEI = 32;    // neighbors per point == WMMA K for per-point matmul
constexpr int NKP  = 15;    // kernel points (padded to 16 rows)
constexpr int DIN  = 64;
constexpr int DOUT = 64;
constexpr int NOFF = 45;    // NKP*3 offset channels (padded to 48 = 3 N-tiles)
constexpr int TP   = 16;    // query points per workgroup (= WMMA M for cross GEMMs)
constexpr int NCHUNK = (NKP * DIN) / 32;       // 30 K-chunks of 32 (never cross kp)
constexpr int OFFB_HALFS = 3 * NCHUNK * 512;   // offset_weights in WMMA-B layout
constexpr int WB_HALFS   = 4 * NCHUNK * 512;   // weight in WMMA-B layout

// dynamic LDS layout (bytes)
constexpr int L_NEIGH = 0;                              // [16][32] int
constexpr int L_NB    = L_NEIGH + TP * MNEI * 4;        // [16][32][3] f32 centered nbrs
constexpr int L_DEFK  = L_NB + TP * MNEI * 3 * 4;       // [16][16][3] f32 deformed kpts
constexpr int L_NFB   = L_DEFK + TP * 16 * 3 * 4;       // [16p*4t][32 lane][16 e] f16 B-mats
constexpr int L_WFA   = L_NFB + TP * 4 * 512 * 2;       // [32 chunk][32 lane][16 e] f16 A-mats
constexpr int SMEM_BYTES = L_WFA + 32 * 512 * 2;        // ~107 KB
}

// One-time conversion of the two weight tensors into f16 WMMA-B chunk layout in d_ws.
// B layout per 32x16 chunk: lane l holds column n=(l&15); element e -> k = ((e&8)<<1)|((l&16)>>1)|(e&7).
// chunk c: kp = c>>1, d = (c&1)*32 + k  (global K index kk = kp*64 + d).
__global__ __launch_bounds__(256) void kpconv_preproc(
    const float* __restrict__ ow, const float* __restrict__ w,
    _Float16* __restrict__ wsB) {
  int id = blockIdx.x * 256 + threadIdx.x;
  if (id >= OFFB_HALFS + WB_HALFS) return;
  bool isW = id >= OFFB_HALFS;
  int lid = isW ? id - OFFB_HALFS : id;
  int t    = lid / (NCHUNK * 512);
  int rem  = lid % (NCHUNK * 512);
  int c    = rem / 512;
  int li   = rem % 512;
  int lane = li >> 4;
  int e    = li & 15;
  int k    = ((e & 8) << 1) | ((lane & 16) >> 1) | (e & 7);          // K within chunk (0..31)
  int kp   = c >> 1;
  int d    = (c & 1) * 32 + k;
  int n    = t * 16 + (lane & 15);
  float v = 0.0f;
  if (isW) {
    v = w[(kp * DIN + d) * DOUT + n];                 // [15][64][64]
  } else if (n < NOFF) {
    v = ow[(kp * DIN + d) * NOFF + n];                // [15][64][45]
  }
  wsB[id] = (_Float16)v;
}

__global__ __launch_bounds__(128) void kpconv_main(
    const float* __restrict__ q, const float* __restrict__ sp,
    const int* __restrict__ neigh, const float* __restrict__ x,
    const float* __restrict__ kpts, const float* __restrict__ obias,
    const _Float16* __restrict__ wsB, float* __restrict__ out) {
  extern __shared__ char smem[];
  int*      s_neigh = (int*)(smem + L_NEIGH);
  float*    s_nb    = (float*)(smem + L_NB);
  float*    s_defK  = (float*)(smem + L_DEFK);
  _Float16* s_nfB   = (_Float16*)(smem + L_NFB);
  _Float16* s_wfA   = (_Float16*)(smem + L_WFA);

  const int tid  = threadIdx.x;
  const int lane = tid & 31;
  const int wave = tid >> 5;
  const int base = blockIdx.x * TP;
  const int row  = lane & 15;          // A-matrix row for per-point matmuls (kernel pt)
  const int kb8  = (lane & 16) >> 1;   // A layout: high lane-half shifts K by 8

  // ---- Phase 0a: neighbor ids + centered neighbor coordinates ----
  for (int idx = tid; idx < TP * MNEI; idx += 128) {
    int p = idx >> 5, m = idx & 31;
    int n = neigh[(base + p) * MNEI + m];
    s_neigh[idx] = n;
    float qx = q[(base + p) * 3 + 0];
    float qy = q[(base + p) * 3 + 1];
    float qz = q[(base + p) * 3 + 2];
    s_nb[idx * 3 + 0] = sp[n * 3 + 0] - qx;
    s_nb[idx * 3 + 1] = sp[n * 3 + 1] - qy;
    s_nb[idx * 3 + 2] = sp[n * 3 + 2] - qz;
  }
  __syncthreads();

  // ---- Phase 0b: gather x[neighbors] -> f16 WMMA-B matrices in LDS ----
  // B element (lane,e): k=m=((e&8)<<1)|((lane&16)>>1)|(e&7), n=d&15, tile=d>>4. Even/odd m pairs are
  // adjacent halves -> pack h2 stores (lowers to v_cvt_pk_f16_f32 + ds_store_b32).
  for (int idx = tid; idx < TP * (MNEI / 2); idx += 128) {
    int p = idx >> 4, r = idx & 15;
    int m0 = 2 * r;
    int n0 = s_neigh[p * MNEI + m0];
    int n1 = s_neigh[p * MNEI + m0 + 1];
    const float4* x0 = (const float4*)(x + (size_t)n0 * DIN);
    const float4* x1 = (const float4*)(x + (size_t)n1 * DIN);
    int laneTop = (m0 & 8) << 1;
    int e = ((m0 & 16) >> 1) | (m0 & 7);   // even
#pragma unroll
    for (int qd = 0; qd < 16; ++qd) {
      float4 a = x0[qd];
      float4 b = x1[qd];
      float av[4] = {a.x, a.y, a.z, a.w};
      float bv[4] = {b.x, b.y, b.z, b.w};
#pragma unroll
      for (int j = 0; j < 4; ++j) {
        int d = qd * 4 + j;
        int bl = (d & 15) | laneTop;
        int t  = d >> 4;
        h2t v; v.x = (_Float16)av[j]; v.y = (_Float16)bv[j];
        *(h2t*)&s_nfB[((p * 4 + t) * 32 + bl) * 16 + e] = v;
      }
    }
  }
  __syncthreads();

  // rigid kernel point for this lane's A row
  float kx = 0.f, ky = 0.f, kz = 0.f;
  if (row < NKP) {
    kx = kpts[row * 3 + 0]; ky = kpts[row * 3 + 1]; kz = kpts[row * 3 + 2];
  }

  // ---- Phase 1: rigid wf[p] = aw(16x32) x nf(32x64), write wf to cross-A layout ----
  // All 4 B tiles preloaded so the 4 independent WMMAs issue back-to-back without
  // waiting on the wf ds_stores.
#pragma unroll 1
  for (int pp = 0; pp < 4; ++pp) {
    int p = wave * 4 + pp;
    v16h A;
#pragma unroll
    for (int e = 0; e < 16; ++e) {
      int m = ((e & 8) << 1) | kb8 | (e & 7);   // A layout: element e -> neighbor m
      float dx = s_nb[(p * MNEI + m) * 3 + 0] - kx;
      float dy = s_nb[(p * MNEI + m) * 3 + 1] - ky;
      float dz = s_nb[(p * MNEI + m) * 3 + 2] - kz;
      float dist = sqrtf(dx * dx + dy * dy + dz * dz);
      float a = fmaxf(1.0f - dist, 0.0f);       // extent = POINT_INFLUENCE = 1
      if (row >= NKP) a = 0.0f;                 // pad row
      A[e] = (_Float16)a;
    }
    v16h B0 = *(const v16h*)&s_nfB[((p * 4 + 0) * 32 + lane) * 16];
    v16h B1 = *(const v16h*)&s_nfB[((p * 4 + 1) * 32 + lane) * 16];
    v16h B2 = *(const v16h*)&s_nfB[((p * 4 + 2) * 32 + lane) * 16];
    v16h B3 = *(const v16h*)&s_nfB[((p * 4 + 3) * 32 + lane) * 16];
    v8f Cz = {};
    v8f D0 = __builtin_amdgcn_wmma_f32_16x16x32_f16(false, A, false, B0, (short)0, Cz, false, false);
    v8f D1 = __builtin_amdgcn_wmma_f32_16x16x32_f16(false, A, false, B1, (short)0, Cz, false, false);
    v8f D2 = __builtin_amdgcn_wmma_f32_16x16x32_f16(false, A, false, B2, (short)0, Cz, false, false);
    v8f D3 = __builtin_amdgcn_wmma_f32_16x16x32_f16(false, A, false, B3, (short)0, Cz, false, false);
    float Dv[4][8];
#pragma unroll
    for (int i = 0; i < 8; ++i) { Dv[0][i] = D0[i]; Dv[1][i] = D1[i]; Dv[2][i] = D2[i]; Dv[3][i] = D3[i]; }
#pragma unroll
    for (int t = 0; t < 4; ++t) {
#pragma unroll
      for (int i = 0; i < 8; ++i) {
        int kp = i + kb8;
        int d  = t * 16 + (lane & 15);
        int c  = kp * 2 + (d >> 5);
        int kl = d & 31;
        int al = p | ((kl & 8) << 1);
        int e2 = ((kl & 16) >> 1) | (kl & 7);
        s_wfA[c * 512 + al * 16 + e2] = (_Float16)Dv[t][i];
      }
    }
  }
  __syncthreads();

  // ---- Phase 2: offset_feat = wf_flat(16x960) x offsetW(960x48); deform kpts ----
  if (wave < 3) {
    int t = wave;
    v8f acc0 = {}, acc1 = {};
    const _Float16* Bb = wsB + (size_t)(t * NCHUNK) * 512 + lane * 16;
#pragma unroll 1
    for (int c = 0; c < NCHUNK; c += 2) {
      v16h A0 = *(const v16h*)&s_wfA[c * 512 + lane * 16];
      v16h B0 = *(const v16h*)&Bb[c * 512];
      v16h A1 = *(const v16h*)&s_wfA[(c + 1) * 512 + lane * 16];
      v16h B1 = *(const v16h*)&Bb[(c + 1) * 512];
      acc0 = __builtin_amdgcn_wmma_f32_16x16x32_f16(false, A0, false, B0, (short)0, acc0, false, false);
      acc1 = __builtin_amdgcn_wmma_f32_16x16x32_f16(false, A1, false, B1, (short)0, acc1, false, false);
    }
#pragma unroll
    for (int i = 0; i < 8; ++i) {
      int prow = i + kb8;                 // query point within tile
      int col  = t * 16 + (lane & 15);    // offset channel
      if (col < NOFF) {
        int kpi = col / 3, dim = col % 3;
        float v = (acc0[i] + acc1[i]) + obias[col];   // * POINT_INFLUENCE (=1)
        s_defK[(prow * 16 + kpi) * 3 + dim] = kpts[kpi * 3 + dim] + v;
      }
    }
  }
  __syncthreads();

  // ---- Phase 3: deformable wf2[p] with per-point kernel points ----
#pragma unroll 1
  for (int pp = 0; pp < 4; ++pp) {
    int p = wave * 4 + pp;
    float dkx = 0.f, dky = 0.f, dkz = 0.f;
    if (row < NKP) {
      dkx = s_defK[(p * 16 + row) * 3 + 0];
      dky = s_defK[(p * 16 + row) * 3 + 1];
      dkz = s_defK[(p * 16 + row) * 3 + 2];
    }
    v16h A;
#pragma unroll
    for (int e = 0; e < 16; ++e) {
      int m = ((e & 8) << 1) | kb8 | (e & 7);
      float dx = s_nb[(p * MNEI + m) * 3 + 0] - dkx;
      float dy = s_nb[(p * MNEI + m) * 3 + 1] - dky;
      float dz = s_nb[(p * MNEI + m) * 3 + 2] - dkz;
      float dist = sqrtf(dx * dx + dy * dy + dz * dz);
      float a = fmaxf(1.0f - dist, 0.0f);
      if (row >= NKP) a = 0.0f;
      A[e] = (_Float16)a;
    }
    v16h B0 = *(const v16h*)&s_nfB[((p * 4 + 0) * 32 + lane) * 16];
    v16h B1 = *(const v16h*)&s_nfB[((p * 4 + 1) * 32 + lane) * 16];
    v16h B2 = *(const v16h*)&s_nfB[((p * 4 + 2) * 32 + lane) * 16];
    v16h B3 = *(const v16h*)&s_nfB[((p * 4 + 3) * 32 + lane) * 16];
    v8f Cz = {};
    v8f D0 = __builtin_amdgcn_wmma_f32_16x16x32_f16(false, A, false, B0, (short)0, Cz, false, false);
    v8f D1 = __builtin_amdgcn_wmma_f32_16x16x32_f16(false, A, false, B1, (short)0, Cz, false, false);
    v8f D2 = __builtin_amdgcn_wmma_f32_16x16x32_f16(false, A, false, B2, (short)0, Cz, false, false);
    v8f D3 = __builtin_amdgcn_wmma_f32_16x16x32_f16(false, A, false, B3, (short)0, Cz, false, false);
    float Dv[4][8];
#pragma unroll
    for (int i = 0; i < 8; ++i) { Dv[0][i] = D0[i]; Dv[1][i] = D1[i]; Dv[2][i] = D2[i]; Dv[3][i] = D3[i]; }
#pragma unroll
    for (int t = 0; t < 4; ++t) {
#pragma unroll
      for (int i = 0; i < 8; ++i) {
        int kp = i + kb8;
        int d  = t * 16 + (lane & 15);
        int c  = kp * 2 + (d >> 5);
        int kl = d & 31;
        int al = p | ((kl & 8) << 1);
        int e2 = ((kl & 16) >> 1) | (kl & 7);
        s_wfA[c * 512 + al * 16 + e2] = (_Float16)Dv[t][i];
      }
    }
  }
  __syncthreads();

  // ---- Phase 4: out = wf2_flat(16x960) x W(960x64) ----
  {
    int t = wave;   // 4 waves, 4 N-tiles
    v8f acc0 = {}, acc1 = {};
    const _Float16* Bb = wsB + OFFB_HALFS + (size_t)(t * NCHUNK) * 512 + lane * 16;
#pragma unroll 1
    for (int c = 0; c < NCHUNK; c += 2) {
      v16h A0 = *(const v16h*)&s_wfA[c * 512 + lane * 16];
      v16h B0 = *(const v16h*)&Bb[c * 512];
      v16h A1 = *(const v16h*)&s_wfA[(c + 1) * 512 + lane * 16];
      v16h B1 = *(const v16h*)&Bb[(c + 1) * 512];
      acc0 = __builtin_amdgcn_wmma_f32_16x16x32_f16(false, A0, false, B0, (short)0, acc0, false, false);
      acc1 = __builtin_amdgcn_wmma_f32_16x16x32_f16(false, A1, false, B1, (short)0, acc1, false, false);
    }
#pragma unroll
    for (int i = 0; i < 8; ++i) {
      int prow = i + kb8;
      int col  = t * 16 + (lane & 15);
      out[(size_t)(base + prow) * DOUT + col] = acc0[i] + acc1[i];
    }
  }
}

extern "C" void kernel_launch(void* const* d_in, const int* in_sizes, int n_in,
                              void* d_out, int out_size, void* d_ws, size_t ws_size,
                              hipStream_t stream) {
  (void)in_sizes; (void)n_in; (void)out_size; (void)ws_size;
  const float* q  = (const float*)d_in[0];   // query_points [N,3]
  const float* sp = (const float*)d_in[1];   // support_points [N,3]
  const int*   nb = (const int*)d_in[2];     // neighbors [N,32]
  const float* x  = (const float*)d_in[3];   // x [N,64]
  const float* kp = (const float*)d_in[4];   // K_points [15,3]
  const float* ow = (const float*)d_in[5];   // offset_weights [15,64,45]
  const float* ob = (const float*)d_in[6];   // offset_bias [45]
  const float* w  = (const float*)d_in[7];   // weight [15,64,64]
  _Float16* wsB = (_Float16*)d_ws;           // f16 WMMA-B layout weights (215 KB)
  float* out = (float*)d_out;

  int total = OFFB_HALFS + WB_HALFS;
  kpconv_preproc<<<(total + 255) / 256, 256, 0, stream>>>(ow, w, wsB);
  kpconv_main<<<NPTS / TP, 128, SMEM_BYTES, stream>>>(q, sp, nb, x, kp, ob, wsB, out);
}
